// Diffusion_57956288692549
// MI455X (gfx1250) — compile-verified, matrix operand order checked by reading, not run on hardware
//
#include <hip/hip_runtime.h>
#include <math.h>

// ---------------------------------------------------------------------------
// MI455X (gfx1250) fused diffusion-ranker.
// Math rewrites: logits = (q@Wk)·te + x  (cond term cancels in softmax);
// attn output = ((Σ_v w·te) + cond) @ Wv.T @ Wp.T;  decoder concat split into
// per-batch hb = ne@Wd1[:, :D].T + bd1  and big GEMM te @ Wd1[:, D:].T done in
// bf16 WMMA (v_wmma_f32_16x16x32_bf16), relu+dot(Wd2) fused into the epilogue.
// Decoder pipelines B tiles into LDS with GLOBAL_LOAD_ASYNC_TO_LDS_B128 +
// double buffering (ASYNCcnt), falling back to sync copies if the builtins
// are unavailable.
// ---------------------------------------------------------------------------

typedef __attribute__((ext_vector_type(16))) __bf16 v16bf;
typedef __attribute__((ext_vector_type(8)))  __bf16 v8bf;
typedef __attribute__((ext_vector_type(8)))  float  v8f;
typedef int vint4 __attribute__((vector_size(16)));   // builtin's b128 unit

#if defined(__HIP_DEVICE_COMPILE__) &&                                        \
    __has_builtin(__builtin_amdgcn_global_load_async_to_lds_b128) &&          \
    __has_builtin(__builtin_amdgcn_s_wait_asynccnt)
#define USE_ASYNC_LDS 1
#else
#define USE_ASYNC_LDS 0
#endif

#define GLB_AS __attribute__((address_space(1)))
#define LDS_AS __attribute__((address_space(3)))

// A/B fragment layout (ISA 7.12.2, 16-bit 16x32): lane L holds row/col L%16;
// lanes 0-15 carry K = {kb..kb+7, kb+16..kb+23}, lanes 16-31 shifted by 8.
static __device__ __forceinline__ v16bf cvt_frag(const float* p0, const float* p1) {
  v16bf r;
#pragma unroll
  for (int i = 0; i < 8; ++i) { r[i] = (__bf16)p0[i]; r[8 + i] = (__bf16)p1[i]; }
  return r;
}
static __device__ __forceinline__ v16bf cvt_frag_add(const float* p0, const float* p1,
                                                     const float* q0, const float* q1) {
  v16bf r;
#pragma unroll
  for (int i = 0; i < 8; ++i) {
    r[i]     = (__bf16)(p0[i] + q0[i]);
    r[8 + i] = (__bf16)(p1[i] + q1[i]);
  }
  return r;
}

// --- sinusoidal timestep embedding: T[b, d] --------------------------------
__global__ void pe_kernel(const int* __restrict__ ts, float* __restrict__ T) {
  int b = blockIdx.x, i = threadIdx.x;              // i in [0,256): d = 2i, 2i+1
  float pos  = (float)ts[b];
  float freq = __expf((float)(2 * i) * (-9.2103403719761836f / 512.0f));
  float arg  = pos * freq;
  T[b * 512 + 2 * i]     = sinf(arg);
  T[b * 512 + 2 * i + 1] = cosf(arg);
}

// --- Wk transpose (so qk = q@Wk reuses the W.T GEMM path) ------------------
__global__ void transpose_kernel(const float* __restrict__ Wk, float* __restrict__ WkT) {
  int idx = blockIdx.x * 256 + threadIdx.x;         // [0, 262144)
  int d = idx >> 9, e = idx & 511;
  WkT[e * 512 + d] = Wk[d * 512 + e];
}

// --- generic small GEMM: out = act((A [+Aadd]) @ W[:, koff:koff+512].T + bias)
// one wave per 16x16 tile; K=512 fully unrolled, compile-time HASADD/ACT so
// the k-loop is a branch-free clause of loads -> v_cvt_pk_bf16 -> v_wmma.
template <bool HASADD, int ACT>
__global__ void __launch_bounds__(256)
gemm_wmma_kernel(const float* __restrict__ A, const float* __restrict__ Aadd, int lda,
                 const float* __restrict__ W, int ldw, int koff,
                 const float* __restrict__ bias,
                 float* __restrict__ out, int ldo, int Mtiles, int Ntiles) {
  int wave = (blockIdx.x * blockDim.x + threadIdx.x) >> 5;
  int lane = threadIdx.x & 31;
  int ntile = wave / Mtiles, mtile = wave - ntile * Mtiles;
  if (ntile >= Ntiles) return;                      // wave-uniform: EXEC stays all-1s
  int nl = lane & 15, hi = lane >> 4;
  int m0 = mtile * 16, n0 = ntile * 16;
  const float* arow = A + (size_t)(m0 + nl) * lda + 8 * hi;
  const float* qrow = HASADD ? (Aadd + (size_t)(m0 + nl) * lda + 8 * hi) : nullptr;
  const float* wrow = W + (size_t)(n0 + nl) * ldw + koff + 8 * hi;
  v8f c = {};
#pragma unroll
  for (int kt = 0; kt < 16; ++kt) {
    const float* ap = arow + kt * 32;
    v16bf a = HASADD ? cvt_frag_add(ap, ap + 16, qrow + kt * 32, qrow + kt * 32 + 16)
                     : cvt_frag(ap, ap + 16);
    const float* wp = wrow + kt * 32;
    v16bf bfr = cvt_frag(wp, wp + 16);
    c = __builtin_amdgcn_wmma_f32_16x16x32_bf16(false, a, false, bfr, (short)0, c,
                                                false, false);
  }
#pragma unroll
  for (int r = 0; r < 8; ++r) {
    int m = m0 + r + 8 * hi, n = n0 + nl;
    float v = c[r] + (bias ? bias[n] : 0.0f);
    if (ACT == 1) v = v / (1.0f + __expf(-v));      // SiLU
    out[(size_t)m * ldo + n] = v;
  }
}

// --- attention logits: logits[b,v] = qk[b]·te[b,v] + x[b,v] ----------------
__global__ void logits_kernel(const float* __restrict__ te, const float* __restrict__ QK,
                              const float* __restrict__ x, float* __restrict__ logits) {
  int wave = (blockIdx.x * blockDim.x + threadIdx.x) >> 5;   // = row, [0, 131072)
  int lane = threadIdx.x & 31;
  int row = wave, b = row >> 10;
  const float* tp = te + (size_t)row * 512;
  const float* qp = QK + b * 512;
  float acc = 0.f;
#pragma unroll
  for (int j = 0; j < 16; ++j) acc += tp[lane + 32 * j] * qp[lane + 32 * j];
#pragma unroll
  for (int m = 16; m >= 1; m >>= 1) acc += __shfl_xor(acc, m, 32);
  if (lane == 0) logits[row] = acc + x[row];
}

// --- row softmax over V=1024 ----------------------------------------------
__global__ void softmax_kernel(const float* __restrict__ logits, float* __restrict__ weight) {
  __shared__ float red[256];
  int b = blockIdx.x, tid = threadIdx.x;
  const float* lp = logits + b * 1024;
  float lv[4];
  float mx = -3.4e38f;
#pragma unroll
  for (int j = 0; j < 4; ++j) { lv[j] = lp[tid + 256 * j]; mx = fmaxf(mx, lv[j]); }
  red[tid] = mx; __syncthreads();
  for (int s = 128; s > 0; s >>= 1) { if (tid < s) red[tid] = fmaxf(red[tid], red[tid + s]); __syncthreads(); }
  mx = red[0]; __syncthreads();
  float sum = 0.f;
#pragma unroll
  for (int j = 0; j < 4; ++j) { lv[j] = __expf(lv[j] - mx); sum += lv[j]; }
  red[tid] = sum; __syncthreads();
  for (int s = 128; s > 0; s >>= 1) { if (tid < s) red[tid] += red[tid + s]; __syncthreads(); }
  float inv = 1.0f / red[0];
#pragma unroll
  for (int j = 0; j < 4; ++j) weight[b * 1024 + tid + 256 * j] = lv[j] * inv;
}

// --- weighted target sum: wsum[b,d] = Σ_v weight[b,v]·te[b,v,d] ------------
__global__ void wsum_kernel(const float* __restrict__ te, const float* __restrict__ weight,
                            float* __restrict__ wsum) {
  int b = blockIdx.x, d = threadIdx.x;              // 512 threads
  const float* wp = weight + b * 1024;
  float acc = 0.f;
  for (int v = 0; v < 1024; ++v)
    acc += wp[v] * te[((size_t)(b * 1024 + v)) * 512 + d];
  wsum[b * 512 + d] = acc;
}

// --- pack Wd1[:, D:] to bf16, fragment-friendly: Bpack[kt][n][32 k] --------
__global__ void pack_wd1r_kernel(const float* __restrict__ Wd1, __bf16* __restrict__ Bpack) {
  int t = blockIdx.x * 256 + threadIdx.x;           // [0, 16384)
  int kt = t >> 10, n = t & 1023;
  const float* src = Wd1 + (size_t)n * 1024 + 512 + kt * 32;
  __bf16* dst = Bpack + ((size_t)(kt * 1024 + n)) * 32;
#pragma unroll
  for (int i = 0; i < 32; ++i) dst[i] = (__bf16)src[i];
}

// issue this thread's 64-byte share of one 16KB B tile (async path)
static __device__ __forceinline__ void stage_issue(__bf16* dst, const __bf16* Bpack,
                                                   int n0, int tid) {
  int kts = tid >> 4, nls = tid & 15;
  const __bf16* g = Bpack + ((size_t)(kts * 1024 + n0 + nls)) * 32;
  __bf16* l = dst + nls * 512 + kts * 32;
#if USE_ASYNC_LDS
  // generic->AS casts via integer detour; LDS generic addr truncates to offset.
  // Builtin signature (per clang diagnostic): (v4i AS1*, v4i AS3*, Imm, Imm).
  GLB_AS vint4* gp = (GLB_AS vint4*)(unsigned long long)(uintptr_t)g;
  LDS_AS vint4* lp = (LDS_AS vint4*)(unsigned int)(uintptr_t)l;
  __builtin_amdgcn_global_load_async_to_lds_b128(gp,     lp,     0, 0);
  __builtin_amdgcn_global_load_async_to_lds_b128(gp + 1, lp + 1, 0, 0);
  __builtin_amdgcn_global_load_async_to_lds_b128(gp + 2, lp + 2, 0, 0);
  __builtin_amdgcn_global_load_async_to_lds_b128(gp + 3, lp + 3, 0, 0);
#else
  const uint4* gq = (const uint4*)g;
  uint4* lq = (uint4*)l;
  lq[0] = gq[0]; lq[1] = gq[1]; lq[2] = gq[2]; lq[3] = gq[3];
#endif
}

// --- fused decoder: p[m] = Σ_n relu((te@Wd1R.T)[m,n] + hb[b,n])·wd2[n] + bd2 + w[m]
// wave = 16-row M block; A frags pinned in VGPRs (te read exactly once);
// B tiles double-buffered through LDS via async copies overlapped with WMMA.
__global__ void __launch_bounds__(256)
decoder_kernel(const float* __restrict__ te, const __bf16* __restrict__ Bpack,
               const float* __restrict__ HB, const float* __restrict__ wd2,
               const float* __restrict__ bd2, const float* __restrict__ weight,
               float* __restrict__ out) {
  __shared__ __align__(16) __bf16 sB[2][16 * 512];  // 2 x 16 KB: [n=16][k=512]
  int wave = threadIdx.x >> 5, lane = threadIdx.x & 31;
  int nl = lane & 15, hi = lane >> 4;
  int m0 = (blockIdx.x * 8 + wave) * 16;            // 16-aligned, never crosses b
  int b = m0 >> 10;

#if USE_ASYNC_LDS
  stage_issue(&sB[0][0], Bpack, 0, threadIdx.x);    // prefetch tile 0 (overlaps A load)
#endif

  v16bf afrag[16];                                  // 16 rows x K=512 in registers
#pragma unroll
  for (int kt = 0; kt < 16; ++kt) {
    const float* ap = te + (size_t)(m0 + nl) * 512 + kt * 32 + 8 * hi;
    afrag[kt] = cvt_frag(ap, ap + 16);
  }
  float psum[8];
#pragma unroll
  for (int r = 0; r < 8; ++r) psum[r] = 0.f;

  for (int t = 0; t < 64; ++t) {
#if USE_ASYNC_LDS
    __builtin_amdgcn_s_wait_asynccnt(0);            // own copies landed in LDS
    __syncthreads();                                // everyone's copies + t-1 compute
    if (t + 1 < 64)                                 // stream next tile while computing
      stage_issue(&sB[(t + 1) & 1][0], Bpack, (t + 1) * 16, threadIdx.x);
#else
    __syncthreads();
    stage_issue(&sB[t & 1][0], Bpack, t * 16, threadIdx.x);
    __syncthreads();
#endif
    const __bf16* bbase = &sB[t & 1][0];
    v8f c = {};
#pragma unroll
    for (int kt = 0; kt < 16; ++kt) {
      const __bf16* bp = bbase + nl * 512 + kt * 32 + 8 * hi;
      v8bf lo = *(const v8bf*)bp;
      v8bf hv = *(const v8bf*)(bp + 16);
      v16bf bf;
#pragma unroll
      for (int i = 0; i < 8; ++i) { bf[i] = lo[i]; bf[8 + i] = hv[i]; }
      c = __builtin_amdgcn_wmma_f32_16x16x32_bf16(false, afrag[kt], false, bf,
                                                  (short)0, c, false, false);
    }
    int n0 = t * 16;
    float hbv = HB[(size_t)b * 1024 + n0 + nl];     // col-dependent only
    float w2v = wd2[n0 + nl];
#pragma unroll
    for (int r = 0; r < 8; ++r)
      psum[r] += fmaxf(c[r] + hbv, 0.0f) * w2v;     // relu + dot(Wd2) fused
  }
#pragma unroll
  for (int r = 0; r < 8; ++r) {                     // reduce 16 col-classes per row
#pragma unroll
    for (int m = 8; m >= 1; m >>= 1) psum[r] += __shfl_xor(psum[r], m, 16);
  }
  if (nl == 0) {
    float b2 = bd2[0];
#pragma unroll
    for (int r = 0; r < 8; ++r) {
      int row = m0 + r + 8 * hi;
      out[row] = psum[r] + b2 + weight[row];
    }
  }
}

extern "C" void kernel_launch(void* const* d_in, const int* in_sizes, int n_in,
                              void* d_out, int out_size, void* d_ws, size_t ws_size,
                              hipStream_t stream) {
  const float* x   = (const float*)d_in[0];
  const int*   ts  = (const int*)  d_in[1];
  const float* qe  = (const float*)d_in[2];
  const float* te  = (const float*)d_in[3];
  const float* Wq  = (const float*)d_in[4];
  const float* Wk  = (const float*)d_in[5];
  const float* Wv  = (const float*)d_in[6];
  const float* Wp  = (const float*)d_in[7];
  const float* bp  = (const float*)d_in[8];
  const float* Wt1 = (const float*)d_in[9];
  const float* bt1 = (const float*)d_in[10];
  const float* Wt2 = (const float*)d_in[11];
  const float* bt2 = (const float*)d_in[12];
  const float* Wd1 = (const float*)d_in[13];
  const float* bd1 = (const float*)d_in[14];
  const float* Wd2 = (const float*)d_in[15];
  const float* bd2 = (const float*)d_in[16];
  float* out = (float*)d_out;

  float* ws      = (float*)d_ws;                    // ~5.8 MB total
  float* T       = ws;                              // [128,512]
  float* H1      = ws + 65536;                      // [128,512]
  float* COND    = ws + 131072;                     // [128,512]
  float* Q       = ws + 196608;                     // [128,512]
  float* QK      = ws + 262144;                     // [128,512]
  float* WkT     = ws + 327680;                     // [512,512]
  float* logitsB = ws + 589824;                     // [128,1024]
  float* weight  = ws + 720896;                     // [128,1024]
  float* wsum    = ws + 851968;                     // [128,512]
  float* Z       = ws + 917504;                     // [128,512]
  float* NE      = ws + 983040;                     // [128,512]
  float* HB      = ws + 1048576;                    // [128,1024]
  __bf16* Bpack  = (__bf16*)(ws + 1179648);         // 512*1024 bf16

  pe_kernel<<<128, 256, 0, stream>>>(ts, T);
  pack_wd1r_kernel<<<64, 256, 0, stream>>>(Wd1, Bpack);
  transpose_kernel<<<1024, 256, 0, stream>>>(Wk, WkT);
  // cond = (silu(T@Wt1.T+bt1))@Wt2.T+bt2
  gemm_wmma_kernel<false, 1><<<32, 256, 0, stream>>>(T, nullptr, 512, Wt1, 512, 0, bt1, H1, 512, 8, 32);
  gemm_wmma_kernel<false, 0><<<32, 256, 0, stream>>>(H1, nullptr, 512, Wt2, 512, 0, bt2, COND, 512, 8, 32);
  // q = (query_emb + cond)@Wq.T ; qk = q@Wk
  gemm_wmma_kernel<true, 0><<<32, 256, 0, stream>>>(qe, COND, 512, Wq, 512, 0, nullptr, Q, 512, 8, 32);
  gemm_wmma_kernel<false, 0><<<32, 256, 0, stream>>>(Q, nullptr, 512, WkT, 512, 0, nullptr, QK, 512, 8, 32);
  // attention (cond-term cancels in softmax)
  logits_kernel<<<16384, 256, 0, stream>>>(te, QK, x, logitsB);
  softmax_kernel<<<128, 256, 0, stream>>>(logitsB, weight);
  wsum_kernel<<<128, 512, 0, stream>>>(te, weight, wsum);
  // new_emb = (wsum+cond)@Wv.T@Wp.T + bp ; hb = new_emb@Wd1[:, :D].T + bd1
  gemm_wmma_kernel<true, 0><<<32, 256, 0, stream>>>(wsum, COND, 512, Wv, 512, 0, nullptr, Z, 512, 8, 32);
  gemm_wmma_kernel<false, 0><<<32, 256, 0, stream>>>(Z, nullptr, 512, Wp, 512, 0, bp, NE, 512, 8, 32);
  gemm_wmma_kernel<false, 0><<<64, 256, 0, stream>>>(NE, nullptr, 512, Wd1, 1024, 0, bd1, HB, 1024, 8, 64);
  // fused decoder (dominant 137 GFLOP GEMM + relu + Wd2 dot + weight add)
  decoder_kernel<<<1024, 256, 0, stream>>>(te, Bpack, HB, Wd2, bd2, weight, out);
}
